// LiquidLayer_47682726920740
// MI455X (gfx1250) — matrix operand order repaired and checked by reading, not compiled, not run on hardware
//
#include <hip/hip_runtime.h>
#include <math.h>

#define B_ 64
#define T_ 512
#define I_ 256
#define H_ 1024
#define K_ 1280            // I_ + H_
#define BH (B_*H_)

typedef __attribute__((ext_vector_type(16))) __bf16 v16bf;
typedef __attribute__((ext_vector_type(8)))  __bf16 v8bf;
typedef __attribute__((ext_vector_type(8)))  float  v8f;

// ---------------- workspace layout (bytes) ----------------
#define WC_OFF    0
#define WC_BYTES  (H_*K_*2)            // packed [W_in|W_h] bf16: 2,621,440
#define XBF_OFF   (WC_OFF + WC_BYTES)
#define XBF_BYTES (B_*T_*I_*2)         // x in bf16: 16,777,216
#define HBF_OFF   (XBF_OFF + XBF_BYTES)
#define HBF_BYTES (2*BH*2)             // double-buffered h bf16: 262,144
#define BIAS_OFF  (HBF_OFF + HBF_BYTES)
#define ITAU_OFF  (BIAS_OFF + H_*4)
#define CNT_OFF   (ITAU_OFF + H_*4)    // 4 unsigned counters (one per M-tile)

#define LDS_BYTES (128*H_*2)           // 128 weight rows x 1024 (h-part only) = 256 KB

// ---------------- setup: pack weights to bf16, bias, 1/tau ----------------
__global__ void pack_w_kernel(const float* __restrict__ Winw,
                              const float* __restrict__ Whw,
                              const float* __restrict__ Winb,
                              const float* __restrict__ Whb,
                              const float* __restrict__ rtau,
                              __bf16* __restrict__ Wc,
                              float* __restrict__ bias,
                              float* __restrict__ itau) {
  int idx = blockIdx.x * blockDim.x + threadIdx.x;       // over H_*K_
  if (idx < H_ * K_) {
    int j = idx / K_;
    int k = idx - j * K_;
    float v = (k < I_) ? Winw[j * I_ + k] : Whw[j * H_ + (k - I_)];
    Wc[idx] = (__bf16)v;
  }
  if (idx < H_) {
    bias[idx] = Winb[idx] + Whb[idx];
    float r  = rtau[idx];
    float sp = (r > 20.0f) ? r : log1pf(expf(r));        // softplus
    itau[idx] = 1.0f / (sp + 0.001f);
  }
}

// ---------------- setup: x (f32) -> bf16 ----------------
__global__ void xcvt_kernel(const float* __restrict__ x, __bf16* __restrict__ xbf) {
  int idx = blockIdx.x * blockDim.x + threadIdx.x;       // over B_*T_*I_
  if (idx < B_ * T_ * I_) xbf[idx] = (__bf16)x[idx];
}

// ---------------- setup: zero h state and barrier counters ----------------
__global__ void init_zero_kernel(__bf16* __restrict__ hbf, unsigned* __restrict__ cnt) {
  int idx = blockIdx.x * blockDim.x + threadIdx.x;
  if (idx < 2 * BH) hbf[idx] = (__bf16)0.0f;
  if (idx < 4) cnt[idx] = 0u;
}

// ---------------- per-M-tile grid barrier (8 workgroups share one counter) ---
__device__ __forceinline__ void mt_barrier(unsigned* cnt, unsigned target) {
  __threadfence();            // release: make h_bf16 stores visible at device scope
  __syncthreads();
  if (threadIdx.x == 0) {
    __hip_atomic_fetch_add(cnt, 1u, __ATOMIC_RELEASE, __HIP_MEMORY_SCOPE_AGENT);
    unsigned v;
    do {
      __builtin_amdgcn_s_sleep(1);
      v = __hip_atomic_load(cnt, __ATOMIC_ACQUIRE, __HIP_MEMORY_SCOPE_AGENT);
    } while (v < target);
  }
  __syncthreads();
  __threadfence();            // acquire: invalidate so next-step h loads see fresh data
}

// A-fragment load: lane holds K+akoff..K+akoff+15 (contiguous, matches B)
__device__ __forceinline__ v16bf load_afrag(const __bf16* p, int akoff) {
  v8bf lo = *(const v8bf*)(p + akoff);
  v8bf hi = *(const v8bf*)(p + 8 + akoff);
  return __builtin_shufflevector(lo, hi,
           0,1,2,3,4,5,6,7,8,9,10,11,12,13,14,15);
}

// ---------------- persistent fused scan kernel ----------------
// Grid: 32 workgroups x 256 threads (8 waves -> 2 waves/SIMD32 for latency
// interleave). Tile map: mt = wg>>3 (M-tile, 16 batch rows), jg = wg&7,
// wave w -> N-tile jt = jg*8+w. Each wave owns one 16x16 output tile.
// x-part B fragments live in registers; h-part B streams from 256 KB LDS;
// h-part A is group-double-buffered (4 tiles/group, loads issued one full
// group ahead with overlapping live ranges); f32 state stays in registers.
__global__ void __launch_bounds__(256, 1)
liquid_scan_kernel(const __bf16* __restrict__ xbf,
                   const __bf16* __restrict__ Wc,
                   const float* __restrict__ bias,
                   const float* __restrict__ itau,
                   __bf16* __restrict__ hbf,      // [2][B_][H_] double buffer
                   unsigned* __restrict__ cnt,    // [4]
                   float* __restrict__ out) {
  extern __shared__ v16bf lds_raw[];
  __bf16* Wlds = (__bf16*)lds_raw;                // [128][H_] bf16 (h-part weights)

  const int wg   = blockIdx.x;                    // 0..31
  const int mt   = wg >> 3;                       // 0..3
  const int jg   = wg & 7;                        // 0..7
  const int tid  = threadIdx.x;
  const int wave = tid >> 5;                      // 0..7
  const int lane = tid & 31;
  const int jt   = jg * 8 + wave;                 // 0..63
  const int j0   = jt * 16;
  const int m0   = mt * 16;

  // ---- stage this WG's 128 weight rows (h-part K only) into LDS: async ----
  {
    // chunk c: row = c>>7, k-offset = (c&127)*8 within the 1024-wide h-part
    for (int c = tid; c < 128 * (H_ / 8); c += 256) {
      const int row = c >> 7;
      const int kc  = (c & 127) * 8;
      const __bf16* gsrc = Wc + (size_t)(jg * 128 + row) * K_ + I_ + kc;
      unsigned ldsoff = (unsigned)c * 16u;        // byte offset in LDS segment
      asm volatile("global_load_async_to_lds_b128 %0, %1, off"
                   :: "v"(ldsoff), "v"(gsrc) : "memory");
    }
    asm volatile("s_wait_asynccnt 0x0" ::: "memory");
  }
  __syncthreads();

  const int ln = lane & 15;
  const int lh = lane >> 4;                       // half-wave select
  const int jj = j0 + ln;                         // output column of this lane
  const float bc = bias[jj];
  const float it = itau[jj];

  const int arow = m0 + ln;                       // A-matrix (batch) row
  const __bf16* xrow  = xbf + (size_t)arow * (T_ * I_);
  const __bf16* bbase = Wlds + (size_t)(wave * 16 + ln) * H_; // h-part B row in LDS
  const int akoff = lh ? 16 : 0;                  // A: lanes 16-31 hold K+16..K+31
  const int bkoff = lh ? 16 : 0;                  // B: lanes 16-31 hold K+16..K+31

  // ---- x-part B fragments: register-resident, loaded once from global ----
  v16bf wbx[8];
#pragma unroll
  for (int kt = 0; kt < 8; ++kt)
    wbx[kt] = *(const v16bf*)(Wc + (size_t)jj * K_ + kt * 32 + bkoff);

  float hs[8];                                    // f32 state tile rows (in regs)
#pragma unroll
  for (int r = 0; r < 8; ++r) hs[r] = 0.0f;

  // software pipeline: preload x A-fragments for t = 0
  v16bf xa[8];
#pragma unroll
  for (int kt = 0; kt < 8; ++kt)
    xa[kt] = load_afrag(xrow + kt * 32, akoff);

  for (int t = 0; t < T_; ++t) {
    const __bf16* hr = hbf + (size_t)(t & 1) * BH + (size_t)arow * H_;
    __bf16*       hw = hbf + (size_t)((t + 1) & 1) * BH;

    // 4 independent accumulator chains (bias folded into chain 0)
    v8f acc[4];
    acc[0] = (v8f){bc, bc, bc, bc, bc, bc, bc, bc};
    acc[1] = (v8f){0.f, 0.f, 0.f, 0.f, 0.f, 0.f, 0.f, 0.f};
    acc[2] = (v8f){0.f, 0.f, 0.f, 0.f, 0.f, 0.f, 0.f, 0.f};
    acc[3] = (v8f){0.f, 0.f, 0.f, 0.f, 0.f, 0.f, 0.f, 0.f};

    // ---- issue first h A-group loads (land while the x-part computes) ----
    v16bf a0 = load_afrag(hr + 0 * 32, akoff);
    v16bf a1 = load_afrag(hr + 1 * 32, akoff);
    v16bf a2 = load_afrag(hr + 2 * 32, akoff);
    v16bf a3 = load_afrag(hr + 3 * 32, akoff);

    // ---- input-projection part: K = 0..255, A and B both in registers ----
#pragma unroll
    for (int kt = 0; kt < 8; ++kt) {
      acc[kt & 3] = __builtin_amdgcn_wmma_f32_16x16x32_bf16(
                        false, xa[kt], false, wbx[kt], (short)0, acc[kt & 3], false, false);
    }

    // ---- recurrent part: 8 groups of 4 k-tiles, group-double-buffered A ----
#pragma unroll
    for (int g = 0; g < 8; ++g) {
      v16bf n0, n1, n2, n3;
      if (g < 7) {                                // next group's A: issued early,
        n0 = load_afrag(hr + (g * 4 + 4) * 32, akoff);   // live across 4 WMMAs
        n1 = load_afrag(hr + (g * 4 + 5) * 32, akoff);
        n2 = load_afrag(hr + (g * 4 + 6) * 32, akoff);
        n3 = load_afrag(hr + (g * 4 + 7) * 32, akoff);
      }
      // current group's B from LDS (batched ahead of the WMMAs)
      v16bf b0 = *(const v16bf*)(bbase + (g * 4 + 0) * 32 + bkoff);
      v16bf b1 = *(const v16bf*)(bbase + (g * 4 + 1) * 32 + bkoff);
      v16bf b2 = *(const v16bf*)(bbase + (g * 4 + 2) * 32 + bkoff);
      v16bf b3 = *(const v16bf*)(bbase + (g * 4 + 3) * 32 + bkoff);

      acc[0] = __builtin_amdgcn_wmma_f32_16x16x32_bf16(
                   false, a0, false, b0, (short)0, acc[0], false, false);
      acc[1] = __builtin_amdgcn_wmma_f32_16x16x32_bf16(
                   false, a1, false, b1, (short)0, acc[1], false, false);
      acc[2] = __builtin_amdgcn_wmma_f32_16x16x32_bf16(
                   false, a2, false, b2, (short)0, acc[2], false, false);
      acc[3] = __builtin_amdgcn_wmma_f32_16x16x32_bf16(
                   false, a3, false, b3, (short)0, acc[3], false, false);

      if (g < 7) { a0 = n0; a1 = n1; a2 = n2; a3 = n3; }
    }

    v8f accs = (acc[0] + acc[1]) + (acc[2] + acc[3]);

    // ---- liquid update (f32, in registers) + publish bf16 state ----
#pragma unroll
    for (int r = 0; r < 8; ++r) {
      float dh = tanhf(accs[r]);                  // bias already in accs
      hs[r] += (dh - hs[r]) * it;
      const int row = m0 + r + 8 * lh;            // C-layout: M = r + 8*(lane>=16)
      hw[(size_t)row * H_ + jj] = (__bf16)hs[r];
    }

    // ---- preload next step's x fragments BEFORE the barrier (hides latency) --
    if (t + 1 < T_) {
      const __bf16* xn = xrow + (size_t)(t + 1) * I_;
      __builtin_prefetch((const void*)(xn + I_), 0, 0);   // prefetch t+2 slice
#pragma unroll
      for (int kt = 0; kt < 8; ++kt)
        xa[kt] = load_afrag(xn + kt * 32, akoff);
    }

    // ---- sync the 8 workgroups sharing this M-tile ----
    mt_barrier(&cnt[mt], 8u * (unsigned)(t + 1));
  }

  // ---- final h (f32) ----
#pragma unroll
  for (int r = 0; r < 8; ++r) {
    const int row = m0 + r + 8 * lh;
    out[(size_t)row * H_ + jj] = hs[r];
  }
}

// ---------------- host launcher ----------------
extern "C" void kernel_launch(void* const* d_in, const int* in_sizes, int n_in,
                              void* d_out, int out_size, void* d_ws, size_t ws_size,
                              hipStream_t stream) {
  (void)in_sizes; (void)n_in; (void)out_size; (void)ws_size;

  const float* x    = (const float*)d_in[0];
  const float* Winw = (const float*)d_in[1];
  const float* Winb = (const float*)d_in[2];
  const float* Whw  = (const float*)d_in[3];
  const float* Whb  = (const float*)d_in[4];
  const float* rtau = (const float*)d_in[5];
  float* out = (float*)d_out;

  char* ws = (char*)d_ws;
  __bf16*   Wc   = (__bf16*)(ws + WC_OFF);
  __bf16*   xbf  = (__bf16*)(ws + XBF_OFF);
  __bf16*   hbf  = (__bf16*)(ws + HBF_OFF);
  float*    bias = (float*)(ws + BIAS_OFF);
  float*    itau = (float*)(ws + ITAU_OFF);
  unsigned* cnt  = (unsigned*)(ws + CNT_OFF);

  // pack weights + bias + 1/tau
  pack_w_kernel<<<(H_ * K_ + 255) / 256, 256, 0, stream>>>(
      Winw, Whw, Winb, Whb, rtau, Wc, bias, itau);
  // x -> bf16
  xcvt_kernel<<<(B_ * T_ * I_ + 255) / 256, 256, 0, stream>>>(x, xbf);
  // zero state + barrier counters (fresh every launch: deterministic)
  init_zero_kernel<<<(2 * BH + 255) / 256, 256, 0, stream>>>(hbf, cnt);
  // persistent fused scan: 32 WGs x 256 threads, 256 KB dynamic LDS
  liquid_scan_kernel<<<32, 256, LDS_BYTES, stream>>>(
      xbf, Wc, bias, itau, hbf, cnt, out);
}
